// PostCom2DR_78683800863340
// MI455X (gfx1250) — matrix operand order repaired and checked by reading, not compiled, not run on hardware
//
#include <hip/hip_runtime.h>
#include <hip/hip_bf16.h>
#include <math.h>

// ---------------------------------------------------------------------------
// PostCom2DR forward for MI455X (gfx1250, wave32, WMMA + TDM).
// Heavy path: fused LSTM (input projection + recurrence) as bf16 WMMA GEMM,
// with the x_t tile streamed to LDS by the Tensor Data Mover (double-buffered).
// ---------------------------------------------------------------------------

typedef __attribute__((ext_vector_type(16))) __bf16 v16bf;
typedef __attribute__((ext_vector_type(8)))  __bf16 v8bf;
typedef __attribute__((ext_vector_type(8)))  float  v8f;
typedef __attribute__((ext_vector_type(4)))  unsigned int u32x4;
typedef __attribute__((ext_vector_type(8)))  int i32x8;
typedef __attribute__((ext_vector_type(4)))  int i32x4;

#define B_     8
#define M_     200
#define NC_    199
#define D_     300
#define H_     80
#define G4H    320      // 4*H gates
#define GSTR   324      // padded LDS stride for gate buffer (bank-conflict free)
#define KDIM   416      // padded K: [0,300)=x, [300,320)=0, [320,400)=h, [400,416)=0
#define KT     13       // K tiles of 32
#define NT     20       // N tiles of 16 (320 gates)
#define NSEQC  1592     // B*NC comment sequences

#if defined(__has_builtin)
#  if __has_builtin(__builtin_amdgcn_tensor_load_to_lds)
#    define USE_TDM 1
#  endif
#endif
#ifndef USE_TDM
#  define USE_TDM 0
#endif

__device__ inline __bf16 f2bf(float f) {
  union { float f; unsigned u; } c; c.f = f;
  unsigned r = c.u + 0x7FFFu + ((c.u >> 16) & 1u);   // round-to-nearest-even
  union { unsigned short s; __bf16 b; } o;
  o.s = (unsigned short)(r >> 16);
  return o.b;
}
__device__ inline float sigf(float x) { return 1.0f / (1.0f + expf(-x)); }

// ---------------------------------------------------------------------------
// Kernel 0: pack combined [Wih ; Whh] weights into bf16 WMMA B-fragments and
// combined biases.  B-fragment k order used here (contiguous per lane half, as in build_a):
//   lanes 0-15:  N = lane,    K = ktile*32 + 0..15   (elem e <-> K offset e)
//   lanes 16-31: N = lane-16, K = ktile*32 + 16..31
// Storage: frag-major [kt][nt][lane][16 bf16] so each lane loads 32B straight.
// ---------------------------------------------------------------------------
#define FRAG_ELEMS (KT * NT * 32 * 16)

__global__ void pack_weights(const float* __restrict__ Wih_p, const float* __restrict__ Whh_p,
                             const float* __restrict__ Wih_c, const float* __restrict__ Whh_c,
                             const float* __restrict__ bih_p, const float* __restrict__ bhh_p,
                             const float* __restrict__ bih_c, const float* __restrict__ bhh_c,
                             __bf16* __restrict__ Bp, __bf16* __restrict__ Bc,
                             float* __restrict__ biasp, float* __restrict__ biasc) {
  int tid = blockIdx.x * blockDim.x + threadIdx.x;
  if (tid < 2 * FRAG_ELEMS) {
    int set  = tid / FRAG_ELEMS;          // 0 = post, 1 = comments
    int e    = tid - set * FRAG_ELEMS;
    int elem = e & 15;                    // bf16 element within lane
    int frag = e >> 4;
    int lane = frag & 31;
    int tile = frag >> 5;                 // kt*NT + nt
    int kt   = tile / NT;
    int nt   = tile - kt * NT;
    int n    = nt * 16 + (lane & 15);
    int k    = kt * 32 + ((lane >> 4) << 4) + elem;
    const float* Wih = set ? Wih_c : Wih_p;
    const float* Whh = set ? Whh_c : Whh_p;
    float v = 0.0f;
    if (k < D_)                   v = Wih[(size_t)n * D_ + k];
    else if (k >= 320 && k < 400) v = Whh[(size_t)n * H_ + (k - 320)];
    (set ? Bc : Bp)[e] = f2bf(v);
  } else if (tid < 2 * FRAG_ELEMS + 2 * G4H) {
    int r = tid - 2 * FRAG_ELEMS;
    if (r < G4H) biasp[r] = bih_p[r] + bhh_p[r];
    else { int j = r - G4H; biasc[j] = bih_c[j] + bhh_c[j]; }
  }
}

// ---------------------------------------------------------------------------
// A-fragment builder (16-bit A 16x32, wave32): per lane-half, the 16 bf16
// elements are one contiguous run of 16 (K base..base+15, base = kt*32+half*16),
// so two 16-byte aligned ds_load_b128 feed the fragment directly.
// ---------------------------------------------------------------------------
__device__ inline v16bf build_a(const __bf16* __restrict__ row, int kt, int half) {
  const int base = kt * 32 + half * 16;
  v8bf lo = *(const v8bf*)(row + base);
  v8bf hi = *(const v8bf*)(row + base + 8);
  v16bf a;
#pragma unroll
  for (int j = 0; j < 8; ++j) { a[j] = lo[j]; a[8 + j] = hi[j]; }
  return a;
}

// ---------------------------------------------------------------------------
// Kernel 1: fused LSTM.  grid.x: 0..99 = comment groups of 16 seqs, 100 = post.
// 640 threads = 20 waves; wave w owns gate columns [16w,16w+16).
// Weights live in VGPRs (13 v16bf fragments/lane) for all 200 timesteps.
// x_t tiles are streamed to LDS by the TDM (double-buffered), then converted
// once to bf16 staging; hazards tracked with TENSORcnt + workgroup barriers.
// ---------------------------------------------------------------------------
__global__ __launch_bounds__(640) void lstm_fused(
    const float* __restrict__ post_emb, const float* __restrict__ comments_emb,
    const __bf16* __restrict__ wsBc, const __bf16* __restrict__ wsBp,
    const float* __restrict__ bias_c, const float* __restrict__ bias_p,
    float* __restrict__ post_out,   // (8,200,80)
    float* __restrict__ hid)        // (8,200,80): row0 = post_h, rows1.. = comment h
{
  __shared__ __bf16 Abuf[16 * KDIM];     // bf16 staging of [x_t | h]
  __shared__ float  xstage[2][16 * D_];  // TDM landing buffers (fp32)
  __shared__ float  gbuf[16 * GSTR];     // gate pre-activations
  __shared__ float  cbuf[16 * H_];       // cell state

  const int g       = blockIdx.x;
  const bool isPost = (g == 100);
  const int tid  = threadIdx.x;
  const int lane = tid & 31;
  const int wave = tid >> 5;          // 0..19
  const int half = lane >> 4;

  for (int i = tid; i < 16 * KDIM; i += 640) Abuf[i] = f2bf(0.0f);
  for (int i = tid; i < 16 * H_;   i += 640) cbuf[i] = 0.0f;
  __syncthreads();

  const v16bf* fragsrc = (const v16bf*)(isPost ? wsBp : wsBc);
  const float* bias    = isPost ? bias_p : bias_c;
  const float* xbase   = isPost ? post_emb : comments_emb;
  const int seq0 = isPost ? 0 : g * 16;
  const int nseq = isPost ? B_ : ((NSEQC - seq0) < 16 ? (NSEQC - seq0) : 16);

#if USE_TDM
  // Issue a 2D TDM load of the (nseq x 300) fp32 tile for timestep tt.
  // D# per CDNA5 ISA sec.8: group0 = {count|flags, lds_addr,
  // global_addr, type=2}; group1 = {data_size=4B, dims/tiles/strides}.
  auto issue_tdm = [&](int tt) {
    unsigned long long ga = (unsigned long long)(uintptr_t)xbase +
        ((unsigned long long)seq0 * M_ * D_ + (unsigned long long)tt * D_) * 4ull;
    unsigned lds = (unsigned)(uintptr_t)&xstage[tt & 1][0];   // LDS byte offset (addr[31:0])
    u32x4 g0 = { 1u,                                          // count=1, user mode
                 lds,
                 (unsigned)ga,
                 (unsigned)((ga >> 32) & 0x01FFFFFFull) | 0x80000000u };  // type=2
    i32x8 g1 = { 0x20000,                                     // data_size = 4B
                 (int)(300u << 16),                           // tensor_dim0[15:0]
                 (int)((unsigned)nseq << 16),                 // dim0 hi | tensor_dim1 lo
                 (int)(300u << 16),                           // dim1 hi | tile_dim0=300
                 16,                                          // tile_dim1=16, tile_dim2=0
                 60000,                                       // tensor_dim0_stride = M_*D_
                 0, 0 };
    i32x4 gz4 = { 0, 0, 0, 0 };
    i32x8 gz8 = { 0, 0, 0, 0, 0, 0, 0, 0 };
    __builtin_amdgcn_tensor_load_to_lds(g0, g1, gz4, gz4, gz8, 0);
  };
  if (wave == 0) issue_tdm(0);
#endif

  // Hoist all weight fragments into registers for the whole time loop.
  v16bf bf[KT];
#pragma unroll
  for (int kt = 0; kt < KT; ++kt) bf[kt] = fragsrc[(kt * NT + wave) * 32 + lane];

  const float bval = bias[wave * 16 + (lane & 15)];
  const int m0 = half * 8;                  // C-tile row base for this lane
  const int ncol = wave * 16 + (lane & 15); // C-tile column

  for (int t = 0; t < M_; ++t) {
#if USE_TDM
    // Prefetch next tile via TDM; wait for current tile; publish with barrier.
    if (wave == 0) {
      if (t + 1 < M_) {
        issue_tdm(t + 1);
        __builtin_amdgcn_s_wait_tensorcnt(1);
      } else {
        __builtin_amdgcn_s_wait_tensorcnt(0);
      }
    }
    __syncthreads();
    for (int i = tid; i < 16 * D_; i += 640) {
      int s = i / D_, d = i - s * D_;
      Abuf[s * KDIM + d] = f2bf(xstage[t & 1][i]);
    }
#else
    for (int i = tid; i < 16 * D_; i += 640) {
      int s = i / D_, d = i - s * D_;
      float v = 0.0f;
      if (s < nseq) {
        const float* src = xbase + (size_t)(seq0 + s) * M_ * D_ + (size_t)t * D_;
        v = src[d];
        if (d == 0 && t + 1 < M_) __builtin_prefetch(src + D_, 0, 1);
      }
      Abuf[s * KDIM + d] = f2bf(v);
    }
#endif
    __syncthreads();

    // GEMM: gates(16x320) = A(16x416) @ Wcomb^T(416x320), bf16 WMMA.
    v8f acc;
#pragma unroll
    for (int r = 0; r < 8; ++r) acc[r] = bval;
#pragma unroll
    for (int kt = 0; kt < KT; ++kt) {
      v16bf a = build_a(Abuf + (lane & 15) * KDIM, kt, half);
      acc = __builtin_amdgcn_wmma_f32_16x16x32_bf16(
          false, a, false, bf[kt], (short)0, acc, false, false);
    }
#pragma unroll
    for (int r = 0; r < 8; ++r) gbuf[(r + m0) * GSTR + ncol] = acc[r];
    __syncthreads();

    // Elementwise gates; write new h (bf16) into Abuf h-region for next step.
    for (int e = tid; e < 16 * H_; e += 640) {
      int s = e / H_, j = e - s * H_;
      float gi = gbuf[s * GSTR + j];
      float gf = gbuf[s * GSTR + H_ + j];
      float gg = gbuf[s * GSTR + 2 * H_ + j];
      float go = gbuf[s * GSTR + 3 * H_ + j];
      float c  = sigf(gf) * cbuf[e] + sigf(gi) * tanhf(gg);
      cbuf[e]  = c;
      float h  = sigf(go) * tanhf(c);
      Abuf[s * KDIM + 320 + j] = f2bf(h);
      if (isPost && s < B_) post_out[((size_t)s * M_ + t) * H_ + j] = h;
      if (t == M_ - 1 && s < nseq) {
        if (isPost) hid[(size_t)s * M_ * H_ + j] = h;
        else {
          int q = seq0 + s, b = q / NC_, nc = q - b * NC_;
          hid[((size_t)b * M_ + 1 + nc) * H_ + j] = h;
        }
      }
    }
    __syncthreads();
  }
}

// ---------------------------------------------------------------------------
// Generic small fp32 GEMM with optional transposes / add / tanh / scale.
// grid: (ceil(N/64), M, 8 batches); one thread per output element.
// ---------------------------------------------------------------------------
__global__ void gemm_f32(const float* __restrict__ A, long long sA, int lda, int tA,
                         const float* __restrict__ Bm, long long sB, int ldb, int tB,
                         const float* __restrict__ Cadd, long long sC,
                         float* __restrict__ D, long long sD,
                         int M, int N, int K, int doTanh, float scale) {
  int n = blockIdx.x * 64 + threadIdx.x;
  int m = blockIdx.y;
  int b = blockIdx.z;
  if (n >= N || m >= M) return;
  const float* Ab = A + (size_t)b * sA;
  const float* Bb = Bm + (size_t)b * sB;
  float acc = 0.0f;
  for (int k = 0; k < K; ++k) {
    float av = tA ? Ab[(size_t)k * lda + m] : Ab[(size_t)m * lda + k];
    float bv = tB ? Bb[(size_t)n * ldb + k] : Bb[(size_t)k * ldb + n];
    acc = fmaf(av, bv, acc);
  }
  acc *= scale;
  if (Cadd) acc += Cadd[(size_t)b * sC + (size_t)m * N + n];
  if (doTanh) acc = tanhf(acc);
  D[(size_t)b * sD + (size_t)m * N + n] = acc;
}

// Row-wise softmax in place. grid: (rows, batch); block 64.
__global__ void softmax_rows(float* __restrict__ X, int cols, int ld, long long bstride) {
  float* row = X + (size_t)blockIdx.y * bstride + (size_t)blockIdx.x * ld;
  __shared__ float red[64];
  int t = threadIdx.x;
  float m = -1e30f;
  for (int j = t; j < cols; j += 64) m = fmaxf(m, row[j]);
  red[t] = m; __syncthreads();
  for (int s = 32; s > 0; s >>= 1) { if (t < s) red[t] = fmaxf(red[t], red[t + s]); __syncthreads(); }
  m = red[0]; __syncthreads();
  float sum = 0.0f;
  for (int j = t; j < cols; j += 64) { float e = expf(row[j] - m); row[j] = e; sum += e; }
  red[t] = sum; __syncthreads();
  for (int s = 32; s > 0; s >>= 1) { if (t < s) red[t] += red[t + s]; __syncthreads(); }
  float inv = 1.0f / red[0];
  __syncthreads();
  for (int j = t; j < cols; j += 64) row[j] *= inv;
}

// out[b][j] = sum_i w[i] * tanh(H[b][i][j])
__global__ void tanh_colreduce(const float* __restrict__ H, long long sH,
                               const float* __restrict__ w,
                               float* __restrict__ out, int sOut, int rows, int cols) {
  int j = blockIdx.x * 64 + threadIdx.x;
  int b = blockIdx.y;
  if (j >= cols) return;
  const float* Hb = H + (size_t)b * sH;
  float s = 0.0f;
  for (int i = 0; i < rows; ++i) s = fmaf(w[i], tanhf(Hb[(size_t)i * cols + j]), s);
  out[(size_t)b * sOut + j] = s;
}

// h_global[b] = concat(aw @ post_o, ac[-1] * C[:,-1]); grid 8, block 160.
__global__ void finalize_sample(const float* __restrict__ u, const float* __restrict__ v,
                                const float* __restrict__ post_out,
                                const float* __restrict__ attn,
                                float* __restrict__ h_global) {
  int b = blockIdx.x, k = threadIdx.x;
  const float* aw = u + (size_t)b * 256;
  const float* ac = v + (size_t)b * 256;
  if (k < H_) {
    const float* po = post_out + (size_t)b * M_ * H_;
    float s = 0.0f;
    for (int j = 0; j < M_; ++j) s = fmaf(aw[j], po[(size_t)j * H_ + k], s);
    h_global[(size_t)b * 192 + k] = s;
  } else if (k < 2 * H_) {
    int kk = k - H_;
    const float* at = attn + (size_t)b * (NC_ * H_);
    h_global[(size_t)b * 192 + H_ + kk] = ac[NC_ - 1] * at[(size_t)(NC_ - 1) * H_ + kk];
  }
}

// 1D conv (pad 1) + ReLU + max over positions. grid 8, block 80 (out channel).
__global__ void conv_relu_max(const float* __restrict__ hid,
                              const float* __restrict__ cw, const float* __restrict__ cb,
                              float* __restrict__ h_local) {
  int b = blockIdx.x, o = threadIdx.x;
  float best = 0.0f;
  for (int p = 0; p < NC_; ++p) {
    float s = cb[o];
#pragma unroll
    for (int tt = 0; tt < 3; ++tt) {
      int pp = p + tt - 1;
      if (pp < 0 || pp >= NC_) continue;
      const float* xp = hid + ((size_t)b * M_ + 1 + pp) * H_;
      const float* w  = cw + (size_t)o * H_ * 3 + tt;
      for (int i = 0; i < H_; ++i) s = fmaf(w[i * 3], xp[i], s);
    }
    float r = s > 0.0f ? s : 0.0f;
    best = r > best ? r : best;
  }
  h_local[(size_t)b * H_ + o] = best;
}

// out = h_en @ Wh.T + bh. 16 threads (8 batches x 2 outputs).
__global__ void final_linear(const float* __restrict__ h_global, const float* __restrict__ h_local,
                             const float* __restrict__ Wh, const float* __restrict__ bh,
                             float* __restrict__ out) {
  int t = threadIdx.x;
  if (t >= 16) return;
  int b = t >> 1, r = t & 1;
  float s = bh[r];
  for (int k = 0; k < 160; ++k) s = fmaf(h_global[(size_t)b * 192 + k], Wh[r * 240 + k], s);
  for (int k = 0; k < 80;  ++k) s = fmaf(h_local[(size_t)b * 80 + k],  Wh[r * 240 + 160 + k], s);
  out[b * 2 + r] = s;
}

// ---------------------------------------------------------------------------
extern "C" void kernel_launch(void* const* d_in, const int* in_sizes, int n_in,
                              void* d_out, int out_size, void* d_ws, size_t ws_size,
                              hipStream_t stream) {
  (void)in_sizes; (void)n_in; (void)out_size; (void)ws_size;
  const float* post_emb     = (const float*)d_in[0];
  const float* comments_emb = (const float*)d_in[1];
  const float* adj   = (const float*)d_in[2];
  const float* W0    = (const float*)d_in[3];
  const float* W1    = (const float*)d_in[4];
  const float* Wcw   = (const float*)d_in[5];
  const float* Ww    = (const float*)d_in[6];
  const float* Wc    = (const float*)d_in[7];
  const float* Whw   = (const float*)d_in[8];
  const float* Whc   = (const float*)d_in[9];
  const float* Wh    = (const float*)d_in[10];
  const float* bh    = (const float*)d_in[11];
  const float* Wih_p = (const float*)d_in[12];
  const float* Whh_p = (const float*)d_in[13];
  const float* bih_p = (const float*)d_in[14];
  const float* bhh_p = (const float*)d_in[15];
  const float* Wih_c = (const float*)d_in[16];
  const float* Whh_c = (const float*)d_in[17];
  const float* bih_c = (const float*)d_in[18];
  const float* bhh_c = (const float*)d_in[19];
  const float* conv_w = (const float*)d_in[20];
  const float* conv_b = (const float*)d_in[21];
  float* out = (float*)d_out;

  // Workspace allocator (float units, 32B-aligned slots).
  float* ws = (float*)d_ws;
  size_t off = 0;
  auto alloc = [&](size_t nf) { float* p = ws + off; off += (nf + 7) & ~(size_t)7; return p; };

  __bf16* wsBp = (__bf16*)alloc(FRAG_ELEMS / 2);   // bf16 frags, post
  __bf16* wsBc = (__bf16*)alloc(FRAG_ELEMS / 2);   // bf16 frags, comments
  float* bias_p = alloc(G4H);
  float* bias_c = alloc(G4H);
  float* post_out = alloc((size_t)B_ * M_ * H_);   // ys of post LSTM
  float* hid      = alloc((size_t)B_ * M_ * H_);   // [post_h ; comment hs]
  float* T0  = alloc((size_t)B_ * M_ * H_);        // reused for T1
  float* H1  = alloc((size_t)B_ * M_ * H_);
  float* H2  = alloc((size_t)B_ * M_ * H_);
  float* S   = alloc((size_t)B_ * 39616);          // 199x199 attn scores
  float* attn = alloc((size_t)B_ * NC_ * H_);      // C = attn.T
  float* P   = alloc((size_t)B_ * H_ * M_);
  float* F   = alloc((size_t)B_ * NC_ * M_);
  float* G1  = alloc((size_t)B_ * H_ * M_);
  float* E1  = alloc((size_t)B_ * M_ * M_);        // becomes HcPre
  float* G2  = alloc((size_t)B_ * H_ * NC_);
  float* E2  = alloc((size_t)B_ * M_ * NC_);       // becomes HwPre
  float* u   = alloc((size_t)B_ * 256);            // -> aw (len 200)
  float* v   = alloc((size_t)B_ * 256);            // -> ac (len 199)
  float* h_global = alloc((size_t)B_ * 192);
  float* h_local  = alloc((size_t)B_ * H_);

  // 0) Pack weights to bf16 WMMA fragments + combined biases.
  {
    int total = 2 * FRAG_ELEMS + 2 * G4H;
    pack_weights<<<(total + 255) / 256, 256, 0, stream>>>(
        Wih_p, Whh_p, Wih_c, Whh_c, bih_p, bhh_p, bih_c, bhh_c,
        wsBp, wsBc, bias_p, bias_c);
  }

  // 1) Fused LSTMs (100 comment groups + 1 post group).
  lstm_fused<<<101, 640, 0, stream>>>(post_emb, comments_emb, wsBc, wsBp,
                                      bias_c, bias_p, post_out, hid);

  auto gemm = [&](const float* A, long long sA, int lda, int tA,
                  const float* Bm, long long sB, int ldb, int tB,
                  const float* Cadd, long long sC,
                  float* D, long long sD,
                  int M, int N, int K, int doTanh, float scale) {
    dim3 grid((N + 63) / 64, M, B_);
    gemm_f32<<<grid, 64, 0, stream>>>(A, sA, lda, tA, Bm, sB, ldb, tB,
                                      Cadd, sC, D, sD, M, N, K, doTanh, scale);
  };
  const long long sHid = (long long)M_ * H_;   // 16000
  const long long sAdj = (long long)M_ * M_;   // 40000
  const float rs300 = 1.0f / sqrtf(300.0f);

  // 2) Graph propagation: H1 = tanh(A @ (hid @ W0)); H2 = tanh(A @ (H1 @ W1))
  gemm(hid, sHid, H_, 0,  W0, 0, H_, 0,  nullptr, 0,  T0, sHid, M_, H_, H_, 0, 1.0f);
  gemm(adj, sAdj, M_, 0,  T0, sHid, H_, 0,  nullptr, 0,  H1, sHid, M_, H_, M_, 1, 1.0f);
  gemm(H1, sHid, H_, 0,   W1, 0, H_, 0,  nullptr, 0,  T0, sHid, M_, H_, H_, 0, 1.0f);
  gemm(adj, sAdj, M_, 0,  T0, sHid, H_, 0,  nullptr, 0,  H2, sHid, M_, H_, M_, 1, 1.0f);

  // 3) Self-attention over Xc = H2[1:]: S = Xc@Xc.T/sqrt(300); softmax; attn = S@Xc
  gemm(H2 + H_, sHid, H_, 0,  H2 + H_, sHid, H_, 1,  nullptr, 0,  S, 39616, NC_, NC_, H_, 0, rs300);
  softmax_rows<<<dim3(NC_, B_), 64, 0, stream>>>(S, NC_, NC_, 39616);
  gemm(S, 39616, NC_, 0,  H2 + H_, sHid, H_, 0,  nullptr, 0,  attn, (long long)NC_ * H_, NC_, H_, NC_, 0, 1.0f);

  // 4) Co-attention block (W_dash = post_out[b].T via transposed indexing).
  gemm(Wcw, 0, H_, 0,  post_out, sHid, H_, 1,  nullptr, 0,  P, (long long)H_ * M_, H_, M_, H_, 0, 1.0f);            // P = Wcw@W'
  gemm(attn, (long long)NC_ * H_, H_, 0,  P, (long long)H_ * M_, M_, 0,  nullptr, 0,
       F, (long long)NC_ * M_, NC_, M_, H_, 1, 1.0f);                                                               // F = tanh(C.T@P)
  gemm(attn, (long long)NC_ * H_, H_, 1,  F, (long long)NC_ * M_, M_, 0,  nullptr, 0,
       G1, (long long)H_ * M_, H_, M_, NC_, 0, 1.0f);                                                               // G1 = C@F
  gemm(Ww, 0, H_, 0,  post_out, sHid, H_, 1,  nullptr, 0,  E1, sAdj, M_, M_, H_, 0, 1.0f);                          // E1 = Ww@W'
  gemm(Wc, 0, H_, 0,  G1, (long long)H_ * M_, M_, 0,  E1, sAdj,  E1, sAdj, M_, M_, H_, 0, 1.0f);                    // HcPre = E1 + Wc@G1
  tanh_colreduce<<<dim3((M_ + 63) / 64, B_), 64, 0, stream>>>(E1, sAdj, Whc, u, 256, M_, M_);                       // u = Whc.T tanh(Hc)
  gemm(post_out, sHid, H_, 1,  F, (long long)NC_ * M_, M_, 1,  nullptr, 0,
       G2, (long long)H_ * NC_, H_, NC_, M_, 0, 1.0f);                                                              // G2 = W'@F.T
  gemm(Wc, 0, H_, 0,  attn, (long long)NC_ * H_, H_, 1,  nullptr, 0,
       E2, (long long)M_ * NC_, M_, NC_, H_, 0, 1.0f);                                                              // E2 = Wc@C
  gemm(Ww, 0, H_, 0,  G2, (long long)H_ * NC_, NC_, 0,  E2, (long long)M_ * NC_,
       E2, (long long)M_ * NC_, M_, NC_, H_, 0, 1.0f);                                                              // HwPre = E2 + Ww@G2
  tanh_colreduce<<<dim3((NC_ + 63) / 64, B_), 64, 0, stream>>>(E2, (long long)M_ * NC_, Whw, v, 256, M_, NC_);      // v = Whw.T tanh(Hw)

  // 5) Softmaxes -> aw, ac; then w_bar / c_bar -> h_global.
  softmax_rows<<<dim3(1, B_), 64, 0, stream>>>(u, M_, 0, 256);
  softmax_rows<<<dim3(1, B_), 64, 0, stream>>>(v, NC_, 0, 256);
  finalize_sample<<<B_, 160, 0, stream>>>(u, v, post_out, attn, h_global);

  // 6) Conv + ReLU + max, and final linear head.
  conv_relu_max<<<B_, H_, 0, stream>>>(hid, conv_w, conv_b, h_local);
  final_linear<<<1, 32, 0, stream>>>(h_global, h_local, Wh, bh, out);
}
